// FusedMoE_26603027432076
// MI455X (gfx1250) — compile-verified, matrix operand order checked by reading, not run on hardware
//
#include <hip/hip_runtime.h>
#include <stdint.h>
#include <stddef.h>

#define T_TOK 2048
#define HDIM  1024
#define NEXP  8
#define IDIM  2816

typedef __attribute__((ext_vector_type(16))) __bf16        v16bf;
typedef __attribute__((ext_vector_type(8)))  float         v8f;
typedef __attribute__((ext_vector_type(4)))  unsigned int  v4u;

union FragAB { v16bf v; v4u q[2]; };

__device__ __forceinline__ unsigned short f32_to_bf16_rne(float f) {
  union { float f; unsigned int u; } c; c.f = f;
  unsigned int u = c.u;
  u += 0x7FFFu + ((u >> 16) & 1u);   // round-to-nearest-even
  return (unsigned short)(u >> 16);
}

// CDNA5 async global->LDS copy (ASYNCcnt-tracked, bypasses VGPRs).
// lds_byte_off: wave-relative LDS byte address (= low 32 bits of flat
// address of a __shared__ object per the gfx1250 aperture rules).
__device__ __forceinline__ void async_b128(unsigned lds_byte_off, const void* gsrc) {
  asm volatile("global_load_async_to_lds_b128 %0, %1, off"
               :: "v"(lds_byte_off), "v"(gsrc)
               : "memory");
}

template <int N>
__device__ __forceinline__ void wait_async() {
  asm volatile("s_wait_asynccnt %0" :: "i"(N) : "memory");
}

// ---------------------------------------------------------------------------
// Kernel 0: zero output accumulator + per-expert counts
// ---------------------------------------------------------------------------
__global__ void moe_zero_kernel(float* __restrict__ out, int n, int* __restrict__ counts) {
  const int i = blockIdx.x * blockDim.x + threadIdx.x;
  if (i < n) out[i] = 0.0f;
  if (i < NEXP) counts[i] = 0;
}

// ---------------------------------------------------------------------------
// Kernel 1: router — softmax + top-2 + renormalize + slot assignment
// ---------------------------------------------------------------------------
__global__ void moe_router_kernel(const float* __restrict__ logits,
                                  int* __restrict__ counts,
                                  int* __restrict__ token_map,
                                  float* __restrict__ weight_map) {
  const int t = blockIdx.x * blockDim.x + threadIdx.x;
  if (t >= T_TOK) return;
  float p[NEXP];
  float mx = -1e30f;
  #pragma unroll
  for (int e = 0; e < NEXP; ++e) { p[e] = logits[t * NEXP + e]; mx = fmaxf(mx, p[e]); }
  #pragma unroll
  for (int e = 0; e < NEXP; ++e) p[e] = __expf(p[e] - mx);
  int i0 = 0; float v0 = p[0];
  #pragma unroll
  for (int e = 1; e < NEXP; ++e) if (p[e] > v0) { v0 = p[e]; i0 = e; }
  int i1 = -1; float v1 = -1.0f;
  #pragma unroll
  for (int e = 0; e < NEXP; ++e) if (e != i0 && p[e] > v1) { v1 = p[e]; i1 = e; }
  const float inv = 1.0f / (v0 + v1);
  const int s0 = atomicAdd(&counts[i0], 1);
  token_map[i0 * T_TOK + s0]  = t;
  weight_map[i0 * T_TOK + s0] = v0 * inv;
  const int s1 = atomicAdd(&counts[i1], 1);
  token_map[i1 * T_TOK + s1]  = t;
  weight_map[i1 * T_TOK + s1] = v1 * inv;
}

// ---------------------------------------------------------------------------
// Kernel 2: bulk fp32 -> bf16 cast (vectorized 4-wide, grid-stride)
// ---------------------------------------------------------------------------
__global__ void moe_cast_bf16_kernel(const float4* __restrict__ src,
                                     ushort4* __restrict__ dst, long n4) {
  long i = (long)blockIdx.x * blockDim.x + threadIdx.x;
  const long stride = (long)gridDim.x * blockDim.x;
  for (; i < n4; i += stride) {
    const float4 f = src[i];
    ushort4 o;
    o.x = f32_to_bf16_rne(f.x);
    o.y = f32_to_bf16_rne(f.y);
    o.z = f32_to_bf16_rne(f.z);
    o.w = f32_to_bf16_rne(f.w);
    dst[i] = o;
  }
}

// ---------------------------------------------------------------------------
// Kernel 3: GEMM1 + SiLU*up with async double-buffered LDS pipeline.
//   Block: 256 thr (8 waves), tile BM=32 x BN=64 (gate & up), BK=64.
//   Wave w owns 16x16 subtile (wm, wn) and computes gate+up accumulators.
// ---------------------------------------------------------------------------
__global__ __launch_bounds__(256) void moe_gemm1_silu_kernel(
    const unsigned short* __restrict__ hidb,
    const unsigned short* __restrict__ w13b,
    const int* __restrict__ token_map,
    const int* __restrict__ counts,
    unsigned short* __restrict__ actb) {
  const int e   = blockIdx.z;
  const int cnt = counts[e];
  const int rowStart = blockIdx.y * 32;
  if (rowStart >= cnt) return;               // uniform exit, EXEC stays full
  const int nStart = blockIdx.x * 64;        // column tile within [0, I)

  __shared__ unsigned short As[2][32 * 64];  //  8 KB
  __shared__ unsigned short Bg[2][64 * 64];  // 16 KB
  __shared__ unsigned short Bu[2][64 * 64];  // 16 KB

  const int tid  = threadIdx.x;
  const int lane = tid & 31;
  const int wave = tid >> 5;
  const int hf   = lane >> 4;
  const int l15  = lane & 15;
  const int wm   = (wave >> 2) * 16;
  const int wn   = (wave & 3) * 16;

  // async copy assignments: 1 b128 chunk = 8 bf16
  const int crow = tid >> 3;                 // 0..31
  const int ccol = (tid & 7) * 8;            // 0..56
  const int slot = rowStart + crow;
  const int tok  = (slot < cnt) ? token_map[e * T_TOK + slot] : 0;
  const unsigned short* aSrc  = hidb + (size_t)tok * HDIM + ccol;
  const unsigned short* g0Src = w13b + ((size_t)e * 2 * IDIM + (size_t)(nStart + crow)) * HDIM + ccol;
  const unsigned short* g1Src = g0Src + (size_t)32 * HDIM;
  const unsigned short* u0Src = g0Src + (size_t)IDIM * HDIM;
  const unsigned short* u1Src = g1Src + (size_t)IDIM * HDIM;

  const unsigned asB = (unsigned)(uintptr_t)&As[0][0];
  const unsigned bgB = (unsigned)(uintptr_t)&Bg[0][0];
  const unsigned buB = (unsigned)(uintptr_t)&Bu[0][0];
  const unsigned aOff  = (unsigned)((crow * 64 + ccol) * 2);
  const unsigned b1Off = (unsigned)(((crow + 32) * 64 + ccol) * 2);

  auto issue = [&](int k0, int buf) {
    const unsigned ab = asB + (unsigned)buf * (32 * 64 * 2);
    const unsigned gb = bgB + (unsigned)buf * (64 * 64 * 2);
    const unsigned ub = buB + (unsigned)buf * (64 * 64 * 2);
    async_b128(ab + aOff,  aSrc  + k0);
    async_b128(gb + aOff,  g0Src + k0);
    async_b128(gb + b1Off, g1Src + k0);
    async_b128(ub + aOff,  u0Src + k0);
    async_b128(ub + b1Off, u1Src + k0);
  };

  v8f accg = {};
  v8f accu = {};

  issue(0, 0);
  const int NST = HDIM / 64;                 // 16 stages
  for (int s = 0; s < NST; ++s) {
    const int cur = s & 1;
    if (s + 1 < NST) { issue((s + 1) * 64, 1 - cur); wait_async<5>(); }
    else             { wait_async<0>(); }
    __syncthreads();
    #pragma unroll
    for (int j = 0; j < 2; ++j) {            // two K=32 WMMA steps per stage
      FragAB a, bg, bu;
      const unsigned short* ap = &As[cur][(wm + l15) * 64 + j * 32 + hf * 8];
      const unsigned short* gp = &Bg[cur][(wn + l15) * 64 + j * 32 + hf * 8];
      const unsigned short* up = &Bu[cur][(wn + l15) * 64 + j * 32 + hf * 8];
      a.q[0]  = *(const v4u*)ap;  a.q[1]  = *(const v4u*)(ap + 16);
      bg.q[0] = *(const v4u*)gp;  bg.q[1] = *(const v4u*)(gp + 16);
      bu.q[0] = *(const v4u*)up;  bu.q[1] = *(const v4u*)(up + 16);
      accg = __builtin_amdgcn_wmma_f32_16x16x32_bf16(false, a.v, false, bg.v,
                                                     (short)0, accg, false, false);
      accu = __builtin_amdgcn_wmma_f32_16x16x32_bf16(false, a.v, false, bu.v,
                                                     (short)0, accu, false, false);
    }
    __syncthreads();
  }

  // C/D layout: lane(l15,hf), element r -> M = wm + r + 8*hf, N = wn + l15
  #pragma unroll
  for (int r = 0; r < 8; ++r) {
    const int slotr = rowStart + wm + hf * 8 + r;
    if (slotr < cnt) {
      const float g = accg[r];
      const float u = accu[r];
      const float act = (g / (1.0f + __expf(-g))) * u;   // silu(g) * u
      actb[((size_t)e * T_TOK + slotr) * IDIM + nStart + wn + l15] = f32_to_bf16_rne(act);
    }
  }
}

// ---------------------------------------------------------------------------
// Kernel 4: GEMM2 + weighted scatter-add, async double-buffered pipeline.
//   Block: 256 thr (8 waves), tile BM=32 x BN=64 over H, BK=64.
// ---------------------------------------------------------------------------
__global__ __launch_bounds__(256) void moe_gemm2_scatter_kernel(
    const unsigned short* __restrict__ actb,
    const unsigned short* __restrict__ w2b,
    const int* __restrict__ token_map,
    const float* __restrict__ weight_map,
    const int* __restrict__ counts,
    float* __restrict__ out) {
  const int e   = blockIdx.z;
  const int cnt = counts[e];
  const int rowStart = blockIdx.y * 32;
  if (rowStart >= cnt) return;
  const int hStart = blockIdx.x * 64;

  __shared__ unsigned short As[2][32 * 64];  //  8 KB
  __shared__ unsigned short Bs[2][64 * 64];  // 16 KB

  const int tid  = threadIdx.x;
  const int lane = tid & 31;
  const int wave = tid >> 5;
  const int hf   = lane >> 4;
  const int l15  = lane & 15;
  const int wm   = (wave >> 2) * 16;
  const int wn   = (wave & 3) * 16;

  const int crow = tid >> 3;
  const int ccol = (tid & 7) * 8;
  const unsigned short* aSrc  = actb + ((size_t)e * T_TOK + rowStart + crow) * IDIM + ccol;
  const unsigned short* b0Src = w2b  + ((size_t)e * HDIM + (size_t)(hStart + crow)) * IDIM + ccol;
  const unsigned short* b1Src = b0Src + (size_t)32 * IDIM;

  const unsigned asB = (unsigned)(uintptr_t)&As[0][0];
  const unsigned bsB = (unsigned)(uintptr_t)&Bs[0][0];
  const unsigned aOff  = (unsigned)((crow * 64 + ccol) * 2);
  const unsigned b1Off = (unsigned)(((crow + 32) * 64 + ccol) * 2);

  auto issue = [&](int k0, int buf) {
    const unsigned ab = asB + (unsigned)buf * (32 * 64 * 2);
    const unsigned bb = bsB + (unsigned)buf * (64 * 64 * 2);
    async_b128(ab + aOff,  aSrc  + k0);
    async_b128(bb + aOff,  b0Src + k0);
    async_b128(bb + b1Off, b1Src + k0);
  };

  v8f acc = {};

  issue(0, 0);
  const int NST = IDIM / 64;                 // 44 stages
  for (int s = 0; s < NST; ++s) {
    const int cur = s & 1;
    if (s + 1 < NST) { issue((s + 1) * 64, 1 - cur); wait_async<3>(); }
    else             { wait_async<0>(); }
    __syncthreads();
    #pragma unroll
    for (int j = 0; j < 2; ++j) {
      FragAB a, b;
      const unsigned short* ap = &As[cur][(wm + l15) * 64 + j * 32 + hf * 8];
      const unsigned short* bp = &Bs[cur][(wn + l15) * 64 + j * 32 + hf * 8];
      a.q[0] = *(const v4u*)ap;  a.q[1] = *(const v4u*)(ap + 16);
      b.q[0] = *(const v4u*)bp;  b.q[1] = *(const v4u*)(bp + 16);
      acc = __builtin_amdgcn_wmma_f32_16x16x32_bf16(false, a.v, false, b.v,
                                                    (short)0, acc, false, false);
    }
    __syncthreads();
  }

  #pragma unroll
  for (int r = 0; r < 8; ++r) {
    const int slotr = rowStart + wm + hf * 8 + r;
    if (slotr < cnt) {
      const int   tokr = token_map[e * T_TOK + slotr];
      const float wr   = weight_map[e * T_TOK + slotr];
      atomicAdd(&out[(size_t)tokr * HDIM + hStart + wn + l15], wr * acc[r]);
    }
  }
}

// ---------------------------------------------------------------------------
// Workspace layout (bytes), 256-aligned sections
// ---------------------------------------------------------------------------
static const size_t WS_COUNTS = 0;                                   // 8 ints
static const size_t WS_TOKMAP = 256;                                 // E*T ints
static const size_t WS_WGTMAP = WS_TOKMAP + (size_t)NEXP * T_TOK * 4;
static const size_t WS_HIDB   = WS_WGTMAP + (size_t)NEXP * T_TOK * 4;
static const size_t WS_W13B   = WS_HIDB + (size_t)T_TOK * HDIM * 2;
static const size_t WS_W2B    = WS_W13B + (size_t)NEXP * 2 * IDIM * HDIM * 2;
static const size_t WS_ACTB   = WS_W2B + (size_t)NEXP * HDIM * IDIM * 2;

extern "C" void kernel_launch(void* const* d_in, const int* in_sizes, int n_in,
                              void* d_out, int out_size, void* d_ws, size_t ws_size,
                              hipStream_t stream) {
  const float* hidden = (const float*)d_in[0];   // [T, H]
  const float* logits = (const float*)d_in[1];   // [T, E]
  const float* w13    = (const float*)d_in[2];   // [E, 2I, H]
  const float* w2     = (const float*)d_in[3];   // [E, H, I]
  float* out = (float*)d_out;                    // [T, H]

  char* ws = (char*)d_ws;
  int*            counts     = (int*)(ws + WS_COUNTS);
  int*            token_map  = (int*)(ws + WS_TOKMAP);
  float*          weight_map = (float*)(ws + WS_WGTMAP);
  unsigned short* hidb       = (unsigned short*)(ws + WS_HIDB);
  unsigned short* w13b       = (unsigned short*)(ws + WS_W13B);
  unsigned short* w2b        = (unsigned short*)(ws + WS_W2B);
  unsigned short* actb       = (unsigned short*)(ws + WS_ACTB);

  // 0) zero output + counts
  {
    const int n = T_TOK * HDIM;
    moe_zero_kernel<<<(n + 255) / 256, 256, 0, stream>>>(out, n, counts);
  }
  // 1) router
  moe_router_kernel<<<T_TOK / 256, 256, 0, stream>>>(logits, counts, token_map, weight_map);
  // 2) casts to bf16
  {
    const long nh  = (long)T_TOK * HDIM / 4;
    const long n13 = (long)NEXP * 2 * IDIM * HDIM / 4;
    const long n2  = (long)NEXP * HDIM * IDIM / 4;
    moe_cast_bf16_kernel<<<4096, 256, 0, stream>>>((const float4*)hidden, (ushort4*)hidb, nh);
    moe_cast_bf16_kernel<<<8192, 256, 0, stream>>>((const float4*)w13, (ushort4*)w13b, n13);
    moe_cast_bf16_kernel<<<8192, 256, 0, stream>>>((const float4*)w2, (ushort4*)w2b, n2);
  }
  // 3) GEMM1 + SiLU
  {
    dim3 grid(IDIM / 64, T_TOK / 32, NEXP);
    moe_gemm1_silu_kernel<<<grid, 256, 0, stream>>>(hidb, w13b, token_map, counts, actb);
  }
  // 4) GEMM2 + weighted scatter
  {
    dim3 grid(HDIM / 64, T_TOK / 32, NEXP);
    moe_gemm2_scatter_kernel<<<grid, 256, 0, stream>>>(actb, w2b, token_map, weight_map,
                                                       counts, out);
  }
}
